// FuzzyAND_64338610094619
// MI455X (gfx1250) — matrix operand order repaired and checked by reading, not compiled, not run on hardware
//
#include <hip/hip_runtime.h>
#include <stdint.h>

// ---------------------------------------------------------------------------
// FuzzyAND: out[b,j] = max(0, 1 - sum_i (1-xs[b,i]) * sigmoid(W[i,j]))
// => GEMM M=4096 N=1024 K=1024, bf16 WMMA (f32 accum) on gfx1250, with
//    async global->LDS tile staging (ASYNCcnt) instead of VGPR round-trips.
// ---------------------------------------------------------------------------

typedef __attribute__((ext_vector_type(16))) __bf16 v16bf;
typedef __attribute__((ext_vector_type(8)))  __bf16 v8bf;
typedef __attribute__((ext_vector_type(8)))  float  v8f;
typedef __attribute__((ext_vector_type(4)))  unsigned int u32x4;

#define M_DIM 4096
#define K_DIM 1024
#define N_DIM 1024

__device__ __forceinline__ unsigned short f32_to_bf16(float f) {
  unsigned int u = __float_as_uint(f);
  u += 0x7FFFu + ((u >> 16) & 1u);   // round-to-nearest-even
  return (unsigned short)(u >> 16);
}

__device__ __forceinline__ v16bf cat16(v8bf lo, v8bf hi) {
  return __builtin_shufflevector(lo, hi, 0, 1, 2, 3, 4, 5, 6, 7,
                                         8, 9, 10, 11, 12, 13, 14, 15);
}

// ---- Kernel 1: X' = bf16(1 - xs), row-major [M][K] ------------------------
__global__ void fuzzy_prep_x(const float* __restrict__ xs,
                             unsigned short* __restrict__ xb) {
  int idx = (blockIdx.x * blockDim.x + threadIdx.x) * 8;
  const float4* p = (const float4*)(xs + idx);
  float4 a = p[0];
  float4 b = p[1];
  unsigned int o0 = (unsigned int)f32_to_bf16(1.0f - a.x) |
                    ((unsigned int)f32_to_bf16(1.0f - a.y) << 16);
  unsigned int o1 = (unsigned int)f32_to_bf16(1.0f - a.z) |
                    ((unsigned int)f32_to_bf16(1.0f - a.w) << 16);
  unsigned int o2 = (unsigned int)f32_to_bf16(1.0f - b.x) |
                    ((unsigned int)f32_to_bf16(1.0f - b.y) << 16);
  unsigned int o3 = (unsigned int)f32_to_bf16(1.0f - b.z) |
                    ((unsigned int)f32_to_bf16(1.0f - b.w) << 16);
  u32x4 o = {o0, o1, o2, o3};
  *(u32x4*)(xb + idx) = o;
}

// ---- Kernel 2: Wt = bf16(sigmoid(W))^T, row-major [N][K] ------------------
__global__ void fuzzy_prep_w(const float* __restrict__ w,
                             unsigned short* __restrict__ wt) {
  __shared__ unsigned short tile[32][33];
  const int tx = threadIdx.x;
  const int ty = threadIdx.y;
  const int kBase = blockIdx.y * 32;
  const int nBase = blockIdx.x * 32;
#pragma unroll
  for (int i = 0; i < 4; ++i) {
    int k = kBase + ty + i * 8;
    float x = w[k * N_DIM + nBase + tx];
    float s = 1.0f / (1.0f + __expf(-x));
    tile[ty + i * 8][tx] = f32_to_bf16(s);
  }
  __syncthreads();
#pragma unroll
  for (int i = 0; i < 4; ++i) {
    int n = nBase + ty + i * 8;
    wt[n * K_DIM + kBase + tx] = tile[tx][ty + i * 8];
  }
}

// ---- Kernel 3: tiled WMMA GEMM + Lukasiewicz epilogue ---------------------
// Block tile 128(M) x 128(N), BK=32. 256 threads = 8 wave32, arranged 4x2.
// Each wave: 32x64 macro-tile = 2x4 WMMA accumulators (16x16 each).
// Staging: global_load_async_to_lds_b128, double-buffered LDS, one tile ahead.
#define LDS_STRIDE 40   // 32 bf16 + 8 pad -> 80B rows, conflict-free frag reads

__global__ void __launch_bounds__(256, 2)
fuzzy_gemm(const unsigned short* __restrict__ xb,
           const unsigned short* __restrict__ wtb,
           float* __restrict__ out) {
  __shared__ __align__(16) __bf16 sA[2][128 * LDS_STRIDE];
  __shared__ __align__(16) __bf16 sB[2][128 * LDS_STRIDE];

  const int tid    = threadIdx.x;
  const int lane   = tid & 31;
  const int wid    = tid >> 5;
  const int lane15 = lane & 15;
  const int laneHi = lane >> 4;          // which half-wave

  const int m0 = blockIdx.y * 128;
  const int n0 = blockIdx.x * 128;
  const int mW = (wid & 3) * 32;         // wave M offset inside block tile
  const int nW = (wid >> 2) * 64;        // wave N offset inside block tile

  // Global->LDS staging: 512 16-byte chunks per tile, 2 per thread per matrix.
  const int r0 = tid >> 2;               // rows 0..63
  const int c0 = tid & 3;                // 16B column chunk
  const int r1 = r0 + 64;                // rows 64..127

  const u32x4* gX = (const u32x4*)xb;    // row stride = 1024 bf16 = 128 u32x4
  const u32x4* gW = (const u32x4*)wtb;

  // Per-buffer LDS byte offsets (low 32 bits of the flat pointer are the
  // LDS offset per the gfx1250 aperture mapping).
  unsigned ldsA0[2], ldsA1[2], ldsB0[2], ldsB1[2];
#pragma unroll
  for (int b = 0; b < 2; ++b) {
    ldsA0[b] = (unsigned)(uintptr_t)&sA[b][r0 * LDS_STRIDE + c0 * 8];
    ldsA1[b] = (unsigned)(uintptr_t)&sA[b][r1 * LDS_STRIDE + c0 * 8];
    ldsB0[b] = (unsigned)(uintptr_t)&sB[b][r0 * LDS_STRIDE + c0 * 8];
    ldsB1[b] = (unsigned)(uintptr_t)&sB[b][r1 * LDS_STRIDE + c0 * 8];
  }

  v8f acc[2][4];
  const v8f vzero = {0.f, 0.f, 0.f, 0.f, 0.f, 0.f, 0.f, 0.f};
#pragma unroll
  for (int i = 0; i < 2; ++i)
#pragma unroll
    for (int j = 0; j < 4; ++j) acc[i][j] = vzero;

  // Issue the 4 async copies (A row-chunk x2, B row-chunk x2) for K-slice kt
  // into buffer `buf`. 4 ASYNCcnt ticks per wave per tile.
  auto issue_tile = [&](int kt, int buf) {
    const int kq = kt * 4;               // 32 bf16 = 4 u32x4
    const u32x4* a0 = &gX[(m0 + r0) * 128 + kq + c0];
    const u32x4* a1 = &gX[(m0 + r1) * 128 + kq + c0];
    const u32x4* b0 = &gW[(n0 + r0) * 128 + kq + c0];
    const u32x4* b1 = &gW[(n0 + r1) * 128 + kq + c0];
    asm volatile(
        "global_load_async_to_lds_b128 %0, %4, off\n\t"
        "global_load_async_to_lds_b128 %1, %5, off\n\t"
        "global_load_async_to_lds_b128 %2, %6, off\n\t"
        "global_load_async_to_lds_b128 %3, %7, off"
        :: "v"(ldsA0[buf]), "v"(ldsA1[buf]), "v"(ldsB0[buf]), "v"(ldsB1[buf]),
           "v"(a0), "v"(a1), "v"(b0), "v"(b1)
        : "memory");
  };

  issue_tile(0, 0);

  for (int kt = 0; kt < 32; ++kt) {
    const int buf = kt & 1;
    if (kt + 1 < 32) {
      // Stage the next tile into the other buffer, then wait for this tile's
      // batch (async loads complete in order: <=4 outstanding => tile kt done).
      issue_tile(kt + 1, buf ^ 1);
      asm volatile("s_wait_asynccnt 4" ::: "memory");
    } else {
      asm volatile("s_wait_asynccnt 0" ::: "memory");
    }
    __syncthreads();                     // tile kt visible to all waves

    // A fragments (16x32 bf16): lane<16 holds K{0..7,16..23}, lane>=16
    // holds K{8..15,24..31} for row M=lane15 (ISA 7.12.2).
    v16bf af[2];
#pragma unroll
    for (int i = 0; i < 2; ++i) {
      const __bf16* p = &sA[buf][(mW + i * 16 + lane15) * LDS_STRIDE + laneHi * 8];
      v8bf lo = *(const v8bf*)p;
      v8bf hi = *(const v8bf*)(p + 16);
      af[i] = cat16(lo, hi);
    }
    // B fragments (32x16 bf16): lane<16 holds K{0..7,16..23}, lane>=16
    // holds K{8..15,24..31} for column N=lane15 (same order as A).
    v16bf bfr[4];
#pragma unroll
    for (int j = 0; j < 4; ++j) {
      const __bf16* p = &sB[buf][(nW + j * 16 + lane15) * LDS_STRIDE + laneHi * 8];
      v8bf lo = *(const v8bf*)p;
      v8bf hi = *(const v8bf*)(p + 16);
      bfr[j] = cat16(lo, hi);
    }

#pragma unroll
    for (int i = 0; i < 2; ++i)
#pragma unroll
      for (int j = 0; j < 4; ++j)
        acc[i][j] = __builtin_amdgcn_wmma_f32_16x16x32_bf16(
            false, af[i], false, bfr[j], (short)0, acc[i][j], false, false);

    __syncthreads();                     // all waves done reading buf before
                                         // tile kt+2 overwrites it
  }

  // Epilogue: out = max(1 - s, 0). C/D layout: VGPR v -> M = v (+8 for
  // lanes 16..31), lane15 -> N.
#pragma unroll
  for (int i = 0; i < 2; ++i) {
#pragma unroll
    for (int j = 0; j < 4; ++j) {
      const int col = n0 + nW + j * 16 + lane15;
      const int rowBase = m0 + mW + i * 16 + laneHi * 8;
#pragma unroll
      for (int v = 0; v < 8; ++v) {
        out[(rowBase + v) * N_DIM + col] = fmaxf(1.0f - acc[i][j][v], 0.0f);
      }
    }
  }
}

// ---------------------------------------------------------------------------
extern "C" void kernel_launch(void* const* d_in, const int* in_sizes, int n_in,
                              void* d_out, int out_size, void* d_ws, size_t ws_size,
                              hipStream_t stream) {
  const float* xs = (const float*)d_in[0];   // [4096,1024] f32
  const float* w  = (const float*)d_in[1];   // [1024,1024] f32
  float* out = (float*)d_out;                // [4096,1024] f32

  // Workspace: X' bf16 (8 MiB) then Wt bf16 (2 MiB).
  unsigned short* xb = (unsigned short*)d_ws;
  unsigned short* wt = xb + (size_t)M_DIM * K_DIM;

  fuzzy_prep_x<<<(M_DIM * K_DIM) / (256 * 8), 256, 0, stream>>>(xs, xb);
  fuzzy_prep_w<<<dim3(N_DIM / 32, K_DIM / 32), dim3(32, 8), 0, stream>>>(w, wt);
  fuzzy_gemm<<<dim3(N_DIM / 128, M_DIM / 128), 256, 0, stream>>>(xb, wt, out);
}
